// _EmformerAttention_42494406427091
// MI455X (gfx1250) — matrix operand order repaired and checked by reading, not compiled, not run on hardware
//
#include <hip/hip_runtime.h>
#include <hip/hip_bf16.h>

// ---------------- problem constants ----------------
#define T_    1024
#define R_    64
#define S_    16
#define M_    16
#define B_    8
#define D_    512
#define H_    8
#define DH_   64
#define TQ_   1104            // R + T + S
#define TK_   1104            // M + R + T
#define KK_   512             // K dim of all three GEMMs
#define SCALE_   0.125f       // DH^-0.5
#define NEGINF_  (-100000000.0f)

typedef __bf16 bf16_t;
typedef __attribute__((ext_vector_type(16))) __bf16 v16bf;
typedef __attribute__((ext_vector_type(8)))  __bf16 v8bf;
typedef __attribute__((ext_vector_type(8)))  float  v8f;
typedef __attribute__((ext_vector_type(4)))  int    v4i;

union Frag { v16bf v; v8bf h[2]; };

// ---- CDNA5 async global->LDS copy (ASYNCcnt path), guarded ----
// clang signature (from diagnostic + upstream): (v4i AS1*, v4i AS3*, imm off, imm cpol)
__device__ __forceinline__ void async_copy_b128(const bf16_t* gsrc, bf16_t* lds_dst) {
#if __has_builtin(__builtin_amdgcn_global_load_async_to_lds_b128)
    __builtin_amdgcn_global_load_async_to_lds_b128(
        (__attribute__((address_space(1))) v4i*)gsrc,
        (__attribute__((address_space(3))) v4i*)lds_dst, 0, 0);
#else
    *(v8bf*)lds_dst = *(const v8bf*)gsrc;
#endif
}

__device__ __forceinline__ void async_wait0() {
#if __has_builtin(__builtin_amdgcn_global_load_async_to_lds_b128)
#if __has_builtin(__builtin_amdgcn_s_wait_asynccnt)
    __builtin_amdgcn_s_wait_asynccnt(0);
#else
    asm volatile("s_wait_asynccnt 0" ::: "memory");
#endif
#endif
}

// =====================================================================
// Elementwise packing: build concatenated q_in / kv_in in bf16
// =====================================================================
__global__ __launch_bounds__(256) void pack_qin_kernel(
    const float* __restrict__ utt, const float* __restrict__ rc,
    const float* __restrict__ summ, bf16_t* __restrict__ out)
{
    const size_t total = (size_t)TQ_ * B_ * D_;
    for (size_t idx = (size_t)blockIdx.x * blockDim.x + threadIdx.x;
         idx < total; idx += (size_t)gridDim.x * blockDim.x) {
        size_t tq  = idx / ((size_t)B_ * D_);
        size_t rem = idx % ((size_t)B_ * D_);
        float v;
        if (tq < R_)            v = rc  [ tq              * B_ * D_ + rem];
        else if (tq < R_ + T_)  v = utt [(tq - R_)        * B_ * D_ + rem];
        else                    v = summ[(tq - R_ - T_)   * B_ * D_ + rem];
        out[idx] = (bf16_t)v;
    }
}

__global__ __launch_bounds__(256) void pack_kvin_kernel(
    const float* __restrict__ mems, const float* __restrict__ rc,
    const float* __restrict__ utt, bf16_t* __restrict__ out)
{
    const size_t total = (size_t)TK_ * B_ * D_;
    for (size_t idx = (size_t)blockIdx.x * blockDim.x + threadIdx.x;
         idx < total; idx += (size_t)gridDim.x * blockDim.x) {
        size_t tk  = idx / ((size_t)B_ * D_);
        size_t rem = idx % ((size_t)B_ * D_);
        float v;
        if (tk < M_)            v = mems[ tk              * B_ * D_ + rem];
        else if (tk < M_ + R_)  v = rc  [(tk - M_)        * B_ * D_ + rem];
        else                    v = utt [(tk - M_ - R_)   * B_ * D_ + rem];
        out[idx] = (bf16_t)v;
    }
}

__global__ __launch_bounds__(256) void cvt_f32_bf16_kernel(
    const float* __restrict__ in, bf16_t* __restrict__ out, size_t n)
{
    for (size_t idx = (size_t)blockIdx.x * blockDim.x + threadIdx.x;
         idx < n; idx += (size_t)gridDim.x * blockDim.x)
        out[idx] = (bf16_t)in[idx];
}

// =====================================================================
// bf16 WMMA GEMM (K fixed = 512): C[m][n] = sum_k A[m][k]*W[n][k] + bias[n]
// A: (Mrows x 512) row-major bf16, W: (N x 512) row-major bf16.
// Block = 8 waves; block owns one 16-row M-tile and 8 adjacent N-tiles.
// A strip (16x512, 16KB) is staged once into LDS via async-to-LDS copies
// (shared by all 8 waves); B streams from global (L2-resident weights).
// mode 0: scatter query*SCALE -> qh (B*H, TQ, DH) bf16        (out_a)
// mode 1: scatter kv -> kh (B*H, TK, DH) bf16 (out_a),
//                      vT (B*H, DH, TK) bf16 (out_b)
// mode 2: emformer output epilogue -> outf (d_out layout)
// =====================================================================
__global__ __launch_bounds__(256) void gemm_bf16_wmma_kernel(
    const bf16_t* __restrict__ A, const bf16_t* __restrict__ W,
    const float* __restrict__ bias, int mode,
    float* __restrict__ outf, bf16_t* __restrict__ out_a, bf16_t* __restrict__ out_b)
{
    __shared__ bf16_t lds_a[16 * KK_];          // 16 KB A strip for this M-tile

    const int lane  = threadIdx.x & 31;
    const int wv    = threadIdx.x >> 5;
    const int half  = lane >> 4;
    const int c16   = lane & 15;
    const int ntile = blockIdx.x * 8 + wv;
    const int mtile = blockIdx.y;

    // ---- stage A strip into LDS (async path; 4 x b128 per thread) ----
    {
        const bf16_t* abase = A + (size_t)mtile * 16 * KK_;
        #pragma unroll
        for (int it = 0; it < 4; ++it) {
            int chunk = it * 256 + threadIdx.x;        // 1024 chunks of 8 bf16
            int row   = chunk >> 6;                    // 64 chunks per 512-col row
            int col   = (chunk & 63) * 8;
            async_copy_b128(abase + (size_t)row * KK_ + col,
                            lds_a + row * KK_ + col);
        }
        async_wait0();
        __syncthreads();
    }

    const int n = ntile * 16 + c16;             // W row (output column) for B-frag
    v8f acc = {0.f,0.f,0.f,0.f,0.f,0.f,0.f,0.f};
    const bf16_t* wrow = W + (size_t)n * KK_;

    for (int k = 0; k < KK_; k += 32) {
        Frag a, b;
        const bf16_t* ap = lds_a + c16 * KK_ + k + 8 * half;   // A from LDS
        a.h[0] = *(const v8bf*)(ap);
        a.h[1] = *(const v8bf*)(ap + 16);
        const bf16_t* bp = wrow + k + 8 * half;                // B from global
        b.h[0] = *(const v8bf*)(bp);
        b.h[1] = *(const v8bf*)(bp + 16);
        if (k + 128 < KK_) __builtin_prefetch(wrow + k + 128, 0, 1);
        acc = __builtin_amdgcn_wmma_f32_16x16x32_bf16(
            false, a.v, false, b.v, (short)0, acc, false, false);
    }

    const float bn = bias[n];
    #pragma unroll
    for (int r = 0; r < 8; ++r) {
        const int mr = mtile * 16 + r + 8 * half;     // global output row
        const int nc = n;                             // global output col
        float c = acc[r] + bn;
        if (mode == 0) {
            // query rows are (tq*B + b); cols are h*DH + dh
            int tq = mr / B_, bb = mr % B_;
            int hh = nc / DH_, dh = nc % DH_;
            out_a[((size_t)(bb * H_ + hh) * TQ_ + tq) * DH_ + dh] = (bf16_t)(c * SCALE_);
        } else if (mode == 1) {
            int tk = mr / B_, bb = mr % B_;
            if (nc < D_) {
                int hh = nc / DH_, dh = nc % DH_;
                out_a[((size_t)(bb * H_ + hh) * TK_ + tk) * DH_ + dh] = (bf16_t)c;
            } else {
                int n2 = nc - D_;
                int hh = n2 / DH_, dh = n2 % DH_;
                out_b[((size_t)(bb * H_ + hh) * DH_ + dh) * TK_ + tk] = (bf16_t)c;
            }
        } else {
            int tq = mr / B_, bb = mr % B_;
            if (tq < TQ_ - S_) {
                outf[(size_t)mr * D_ + nc] = c;                       // right context
            } else if (tq < TQ_ - 1) {
                float cc = fminf(fmaxf(c, -10.f), 10.f);              // mems, clipped, drop last
                outf[(size_t)(TQ_ - S_) * B_ * D_ +
                     ((size_t)(tq - (TQ_ - S_)) * B_ + bb) * D_ + nc] = cc;
            }
        }
    }
}

// =====================================================================
// Flash-style attention: one wave per (head, 16-row q tile).
// qh (BH,TQ,DH) bf16 (scale folded), kh (BH,TK,DH) bf16, vT (BH,DH,TK) bf16.
// Output written as bf16 into attn (TQ*B, D) for the final projection.
// =====================================================================
__global__ __launch_bounds__(32) void attn_wmma_kernel(
    const bf16_t* __restrict__ qh, const bf16_t* __restrict__ kh,
    const bf16_t* __restrict__ vt, const unsigned char* __restrict__ amask,
    const int* __restrict__ lengths, bf16_t* __restrict__ attn)
{
    __shared__ bf16_t lds_p[16 * 32];

    const int lane = threadIdx.x & 31;
    const int half = lane >> 4;
    const int c16  = lane & 15;
    const int hd   = blockIdx.x;      // 0..B*H-1  == b*H + h
    const int qt   = blockIdx.y;      // 0..TQ/16-1
    const int q0   = qt * 16;
    const int bb   = hd / H_;
    const int hh   = hd % H_;

    // klengths[b] = lengths[b] + M + (TQ - max(lengths) - S)
    int maxlen = lengths[0];
    #pragma unroll
    for (int i = 1; i < B_; ++i) { int li = lengths[i]; maxlen = (li > maxlen) ? li : maxlen; }
    const int klen = lengths[bb] + M_ + (TQ_ - maxlen - S_);

    // Q fragments (two K=32 slices of DH=64), reused for all key tiles
    Frag qa[2];
    {
        const bf16_t* qrow = qh + ((size_t)hd * TQ_ + (q0 + c16)) * DH_;
        #pragma unroll
        for (int kd = 0; kd < 2; ++kd) {
            const bf16_t* p = qrow + kd * 32 + 8 * half;
            qa[kd].h[0] = *(const v8bf*)(p);
            qa[kd].h[1] = *(const v8bf*)(p + 16);
        }
    }

    float row_m[8], row_l[8];
    #pragma unroll
    for (int r = 0; r < 8; ++r) { row_m[r] = -1e30f; row_l[r] = 0.f; }
    v8f acc[4];
    #pragma unroll
    for (int dt = 0; dt < 4; ++dt)
        acc[dt] = (v8f){0.f,0.f,0.f,0.f,0.f,0.f,0.f,0.f};

    const bf16_t* khead = kh + (size_t)hd * TK_ * DH_;
    const bf16_t* vhead = vt + (size_t)hd * DH_ * TK_;

    for (int kkb = 0; kkb < TK_; kkb += 32) {
        // ---- S = Q @ K^T for 2 adjacent 16-key tiles ----
        v8f s[2];
        #pragma unroll
        for (int kt = 0; kt < 2; ++kt) {
            s[kt] = (v8f){0.f,0.f,0.f,0.f,0.f,0.f,0.f,0.f};
            int krow = kkb + kt * 16 + c16;
            if (krow > TK_ - 1) krow = TK_ - 1;          // clamp tail (masked below)
            const bf16_t* kr = khead + (size_t)krow * DH_;
            #pragma unroll
            for (int kd = 0; kd < 2; ++kd) {
                Frag kb;
                const bf16_t* p = kr + kd * 32 + 8 * half;
                kb.h[0] = *(const v8bf*)(p);
                kb.h[1] = *(const v8bf*)(p + 16);
                s[kt] = __builtin_amdgcn_wmma_f32_16x16x32_bf16(
                    false, qa[kd].v, false, kb.v, (short)0, s[kt], false, false);
            }
        }
        // ---- masks (exact replacement with -1e8, like the reference) ----
        #pragma unroll
        for (int kt = 0; kt < 2; ++kt) {
            int kcol = kkb + kt * 16 + c16;
            int kcl  = (kcol > TK_ - 1) ? (TK_ - 1) : kcol;
            bool colpad = (kcol >= TK_) || (kcol >= klen);
            #pragma unroll
            for (int r = 0; r < 8; ++r) {
                int qrow = q0 + r + 8 * half;
                bool mm = colpad || (amask[(size_t)qrow * TK_ + kcl] != 0);
                if (mm) s[kt][r] = NEGINF_;
            }
        }
        // ---- online softmax (rows live across 16-lane groups) ----
        #pragma unroll
        for (int r = 0; r < 8; ++r) {
            float t = fmaxf(s[0][r], s[1][r]);
            #pragma unroll
            for (int off = 1; off < 16; off <<= 1)
                t = fmaxf(t, __shfl_xor(t, off, 16));
            float mnew = fmaxf(row_m[r], t);
            float corr = __expf(row_m[r] - mnew);
            float p0 = __expf(s[0][r] - mnew);
            float p1 = __expf(s[1][r] - mnew);
            float ps = p0 + p1;
            #pragma unroll
            for (int off = 1; off < 16; off <<= 1)
                ps += __shfl_xor(ps, off, 16);
            row_l[r] = row_l[r] * corr + ps;
            row_m[r] = mnew;
            #pragma unroll
            for (int dt = 0; dt < 4; ++dt) acc[dt][r] *= corr;
            s[0][r] = p0; s[1][r] = p1;
        }
        // ---- relayout P (D-frag f32) -> A-frag bf16 via LDS ----
        __syncthreads();
        #pragma unroll
        for (int kt = 0; kt < 2; ++kt)
            #pragma unroll
            for (int r = 0; r < 8; ++r)
                lds_p[(r + 8 * half) * 32 + kt * 16 + c16] = (bf16_t)s[kt][r];
        __syncthreads();
        Frag pa;
        {
            const bf16_t* p = &lds_p[c16 * 32 + 8 * half];
            pa.h[0] = *(const v8bf*)(p);
            pa.h[1] = *(const v8bf*)(p + 16);
        }
        // ---- O += P @ V  (V pre-transposed: contiguous key chunks) ----
        int kcc = kkb + 8 * half;
        #pragma unroll
        for (int dt = 0; dt < 4; ++dt) {
            Frag vb;
            const bf16_t* p = vhead + (size_t)(dt * 16 + c16) * TK_ + kcc;
            vb.h[0] = *(const v8bf*)(p);
            vb.h[1] = *(const v8bf*)((kcc + 24 <= TK_) ? p + 16 : p);   // tail: p==0 there
            acc[dt] = __builtin_amdgcn_wmma_f32_16x16x32_bf16(
                false, pa.v, false, vb.v, (short)0, acc[dt], false, false);
        }
    }

    // ---- normalize + scatter into attn (TQ*B, D) bf16 ----
    float inv[8];
    #pragma unroll
    for (int r = 0; r < 8; ++r) inv[r] = 1.f / row_l[r];
    #pragma unroll
    for (int dt = 0; dt < 4; ++dt)
        #pragma unroll
        for (int r = 0; r < 8; ++r) {
            int qrow = q0 + r + 8 * half;
            int dd   = hh * DH_ + dt * 16 + c16;
            attn[((size_t)qrow * B_ + bb) * D_ + dd] = (bf16_t)(acc[dt][r] * inv[r]);
        }
}

// =====================================================================
// host launch
// =====================================================================
extern "C" void kernel_launch(void* const* d_in, const int* in_sizes, int n_in,
                              void* d_out, int out_size, void* d_ws, size_t ws_size,
                              hipStream_t stream) {
    (void)in_sizes; (void)n_in; (void)out_size; (void)ws_size;
    const float* utt   = (const float*)d_in[0];
    const float* rc    = (const float*)d_in[1];
    const float* summ  = (const float*)d_in[2];
    const float* mems  = (const float*)d_in[3];
    const float* Wq    = (const float*)d_in[4];
    const float* bq    = (const float*)d_in[5];
    const float* Wkv   = (const float*)d_in[6];
    const float* bkv   = (const float*)d_in[7];
    const float* Wo    = (const float*)d_in[8];
    const float* bo    = (const float*)d_in[9];
    const int*   lens  = (const int*)d_in[10];
    const unsigned char* amask = (const unsigned char*)d_in[11];

    // carve workspace (bf16, 256B-aligned chunks)
    const size_t NQIN = (size_t)TQ_ * B_ * D_;      // 4,521,984
    const size_t NKIN = (size_t)TK_ * B_ * D_;
    const size_t NWQ  = (size_t)D_ * D_;
    const size_t NWKV = (size_t)2 * D_ * D_;
    const size_t NHQ  = (size_t)B_ * H_ * TQ_ * DH_;
    const size_t NHK  = (size_t)B_ * H_ * TK_ * DH_;
    uintptr_t base = ((uintptr_t)d_ws + 255) & ~(uintptr_t)255;
    auto carve = [&](size_t nelem) {
        bf16_t* p = (bf16_t*)base;
        base += ((nelem * sizeof(bf16_t) + 255) & ~(size_t)255);
        return p;
    };
    bf16_t* q_in_bf  = carve(NQIN);
    bf16_t* kv_in_bf = carve(NKIN);
    bf16_t* wq_bf    = carve(NWQ);
    bf16_t* wkv_bf   = carve(NWKV);
    bf16_t* wo_bf    = carve(NWQ);
    bf16_t* qhbf     = carve(NHQ);
    bf16_t* khbf     = carve(NHK);
    bf16_t* vtbf     = carve(NHK);
    bf16_t* attn_bf  = carve(NQIN);

    // 1) pack inputs / weights to bf16
    pack_qin_kernel <<<4096, 256, 0, stream>>>(utt, rc, summ, q_in_bf);
    pack_kvin_kernel<<<4096, 256, 0, stream>>>(mems, rc, utt, kv_in_bf);
    cvt_f32_bf16_kernel<<<1024, 256, 0, stream>>>(Wq,  wq_bf,  NWQ);
    cvt_f32_bf16_kernel<<<2048, 256, 0, stream>>>(Wkv, wkv_bf, NWKV);
    cvt_f32_bf16_kernel<<<1024, 256, 0, stream>>>(Wo,  wo_bf,  NWQ);

    const int MT = (TQ_ * B_) / 16;                 // 552 M-tiles (rows = 8832)

    // 2) query projection -> per-head scaled bf16 Q
    gemm_bf16_wmma_kernel<<<dim3(D_ / 16 / 8, MT), 256, 0, stream>>>(
        q_in_bf, wq_bf, bq, /*mode=*/0, nullptr, qhbf, nullptr);

    // 3) KV projection -> per-head K and transposed V
    gemm_bf16_wmma_kernel<<<dim3(2 * D_ / 16 / 8, MT), 256, 0, stream>>>(
        kv_in_bf, wkv_bf, bkv, /*mode=*/1, nullptr, khbf, vtbf);

    // 4) attention (B*H heads x TQ/16 q-tiles, one wave each)
    attn_wmma_kernel<<<dim3(B_ * H_, TQ_ / 16), 32, 0, stream>>>(
        qhbf, khbf, vtbf, amask, lens, attn_bf);

    // 5) output projection + slice/clip epilogue directly into d_out
    gemm_bf16_wmma_kernel<<<dim3(D_ / 16 / 8, MT), 256, 0, stream>>>(
        attn_bf, wo_bf, bo, /*mode=*/2, (float*)d_out, nullptr, nullptr);
}
